// SparseBasicBlock_37950331027983
// MI455X (gfx1250) — matrix operand order repaired and checked by reading, not compiled, not run on hardware
//
#include <hip/hip_runtime.h>
#include <stdint.h>

// ---------------------------------------------------------------------------
// SparseBasicBlock on MI455X (gfx1250, wave32, WMMA).
//   pre:    x (f32)  -> xbf (bf16)                [memory bound, ~3us]
//   layer1: gather(xbf) x W1 -> BN1 -> ReLU -> mid (bf16)
//   layer2: gather(mid) x W2 -> BN2 -> +x -> ReLU -> out (f32)
// Matmuls: v_wmma_f32_16x16x32_bf16, f32 accumulate; BN/residual in f32.
// Each wave owns 2 site-tiles (32 sites) x 64 out channels (8 accumulators).
// All hot-loop addresses are (scalar base + 32-bit voffset + imm) form:
// per-row byte offsets precomputed once, chunk selectors fold into the
// 24-bit instruction offset of the unrolled loop.
// ---------------------------------------------------------------------------

typedef __attribute__((ext_vector_type(16))) __bf16 v16bf;
typedef __attribute__((ext_vector_type(8)))  float  v8f;

#define BN_EPS      1e-5f
#define N_SITES     200000
#define K_OFF       9
#define CH          64
#define KSTEPS      18                      // 576 contraction / 32 per WMMA
#define WAVES       8
#define MTILES      2                       // site-tiles per wave
#define BLOCK_THREADS (WAVES * 32)
#define N_TILES     (N_SITES / 16)          // 12500 (exact, no tail rows)
#define N_WTILES    (N_TILES / MTILES)      // 6250
// Weights staged in LDS as bf16 pairs in WMMA-B layout:
// dword index = ((kk*2 + half)*64 + d)*8 + q,  pair = K_local (8*half + 2*(q&3) + 16*(q>>2), +1)
#define B_LDS_DWORDS (KSTEPS * 2 * CH * 8)  // 18432 dwords = 73728 B

// Round-to-nearest f32->bf16 pair pack: 2 adds + 1 v_perm_b32.
__device__ __forceinline__ unsigned bfpack2(float a, float b) {
  unsigned ua = __builtin_bit_cast(unsigned, a) + 0x8000u;
  unsigned ub = __builtin_bit_cast(unsigned, b) + 0x8000u;
  // result = { hi16(ub), hi16(ua) }  (even K in [15:0], odd K in [31:16])
  return __builtin_amdgcn_perm(ub, ua, 0x07060302u);
}
__device__ __forceinline__ unsigned short bf1(float a) {
  unsigned ua = __builtin_bit_cast(unsigned, a) + 0x8000u;
  return (unsigned short)(ua >> 16);
}

// Repack W[k][c][d] (f32, row-major) into LDS bf16-pair B-operand layout.
__device__ __forceinline__ void stage_weights(const float* __restrict__ W,
                                              unsigned* __restrict__ lds) {
  for (int i = threadIdx.x; i < B_LDS_DWORDS; i += BLOCK_THREADS) {
    int q    = i & 7;
    int d    = (i >> 3) & 63;
    int half = (i >> 9) & 1;
    int kk   = i >> 10;
    int t = kk * 32 + half * 8 + (q & 3) * 2 + (q >> 2) * 16;    // contraction row (even)
    int k = t >> 6, c = t & 63;             // c even -> (c, c+1) same k
    const float* wp = W + (k * CH + c) * CH + d;
    lds[i] = bfpack2(wp[0], wp[CH]);        // (W[k][c][d], W[k][c+1][d])
  }
}

__device__ __forceinline__ v16bf lds_b16x16(const unsigned* bb) {
  union { uint4 q[2]; v16bf v; } u;
  u.q[0] = *(const uint4*)(bb);
  u.q[1] = *(const uint4*)(bb + 4);
  return u.v;
}
// A gather: scalar base + 32-bit byte offset (+ constant imm) addressing.
__device__ __forceinline__ v16bf gather_a(const char* __restrict__ base,
                                          unsigned boff) {
  union { uint4 q[2]; v16bf v; } u;
  u.q[0] = *(const uint4*)(base + boff);        // K chunk c0..c0+7
  u.q[1] = *(const uint4*)(base + boff + 32u);  // K chunk c0+16..c0+23
  return u.v;
}

#define WMMA_BF16(ACC, A, B) \
  ACC = __builtin_amdgcn_wmma_f32_16x16x32_bf16(false, (A), false, (B), (short)0, (ACC), false, false)

// ---------------------------- pre-pass: f32 -> bf16 ------------------------
__global__ __launch_bounds__(256)
void cvt_bf16(const float* __restrict__ in, unsigned short* __restrict__ outv) {
  const unsigned i = blockIdx.x * 256u + threadIdx.x;  // 8 floats per thread
  if (i >= (N_SITES * CH) / 8) return;
  const float4* p = (const float4*)(in + (size_t)i * 8);
  float4 a = p[0], b = p[1];
  uint4 r;
  r.x = bfpack2(a.x, a.y); r.y = bfpack2(a.z, a.w);
  r.z = bfpack2(b.x, b.y); r.w = bfpack2(b.z, b.w);
  ((uint4*)outv)[i] = r;
}

// ---------------------------- conv layer (templated) -----------------------
template <bool LAYER2>
__global__ __launch_bounds__(BLOCK_THREADS)
void subm_conv(const unsigned short* __restrict__ fin,   // gathered features (bf16)
               const float* __restrict__ xres,           // residual (layer2 only)
               const int* __restrict__ nbr,
               const float* __restrict__ W,
               const float* __restrict__ gam, const float* __restrict__ bet,
               const float* __restrict__ rmean, const float* __restrict__ rvar,
               unsigned short* __restrict__ midout,      // layer1 output (bf16)
               float* __restrict__ fout) {               // layer2 output (f32)
  __shared__ unsigned lwB[B_LDS_DWORDS];
  stage_weights(W, lwB);
  __syncthreads();

  const int lane  = threadIdx.x & 31;
  const int wtile = blockIdx.x * WAVES + (threadIdx.x >> 5);
  if (wtile >= N_WTILES) return;            // wave-uniform: safe around WMMA
  const unsigned row = lane & 15;           // site within tile / out column
  const unsigned g   = lane >> 4;           // lane half selects K sub-chunks

  const char* finb = (const char*)fin;

  unsigned base[MTILES];                    // first site of each tile
  unsigned voff[MTILES][K_OFF];             // gather row byte offsets
#pragma unroll
  for (int m = 0; m < MTILES; ++m) {
    base[m] = (unsigned)(wtile * MTILES + m) * 16u;
#pragma unroll
    for (int k = 0; k < K_OFF; ++k) {
      const unsigned id = (unsigned)nbr[(base[m] + row) * K_OFF + k];
      voff[m][k] = (id << 7) + g * 16u;     // row*128B + lane-half chunk
    }
  }

  v8f acc[MTILES][4];
#pragma unroll
  for (int m = 0; m < MTILES; ++m)
#pragma unroll
    for (int n = 0; n < 4; ++n) acc[m][n] = (v8f){};

#pragma unroll
  for (int kk = 0; kk < KSTEPS; ++kk) {
    const int k = kk >> 1;
    const unsigned csel = (unsigned)(kk & 1) * 64u;  // unroll-time constant

    v16bf A[MTILES];
#pragma unroll
    for (int m = 0; m < MTILES; ++m)
      A[m] = gather_a(finb, voff[m][k] + csel);      // csel folds into imm

    const unsigned* bb = &lwB[((kk * 2 + g) * CH + row) * 8];
    const v16bf B0 = lds_b16x16(bb);
    const v16bf B1 = lds_b16x16(bb + 16 * 8);
    const v16bf B2 = lds_b16x16(bb + 32 * 8);
    const v16bf B3 = lds_b16x16(bb + 48 * 8);
#pragma unroll
    for (int m = 0; m < MTILES; ++m) {      // each B feeds MTILES WMMAs
      WMMA_BF16(acc[m][0], A[m], B0);
      WMMA_BF16(acc[m][1], A[m], B1);
      WMMA_BF16(acc[m][2], A[m], B2);
      WMMA_BF16(acc[m][3], A[m], B3);
    }
  }

  // Epilogue: BN (+residual) (+ReLU). Store offsets affine in j -> j*stride
  // is an unroll-time constant folded into the instruction offset.
  char* outb = LAYER2 ? (char*)fout : (char*)midout;
  const char* resb = (const char*)xres;
#pragma unroll
  for (int nt = 0; nt < 4; ++nt) {
    const unsigned d = nt * 16u + row;      // this lane's output channel
    const float inv  = gam[d] * rsqrtf(rvar[d] + BN_EPS);
    const float beta = bet[d] - rmean[d] * inv;
#pragma unroll
    for (int m = 0; m < MTILES; ++m) {
      const unsigned e0 = (base[m] + g * 8u) * CH + d;  // first element index
#pragma unroll
      for (int j = 0; j < 8; ++j) {         // C/D layout: VGPR j -> row j + g*8
        const float v = acc[m][nt][j] * inv + beta;
        if (LAYER2) {
          const float r = *(const float*)(resb + (size_t)(e0 * 4u) + j * (CH * 4));
          *(float*)(outb + (size_t)(e0 * 4u) + j * (CH * 4)) = fmaxf(v + r, 0.0f);
        } else {
          *(unsigned short*)(outb + (size_t)(e0 * 2u) + j * (CH * 2)) =
              bf1(fmaxf(v, 0.0f));
        }
      }
    }
  }
}

// ---------------------------- launcher -------------------------------------
extern "C" void kernel_launch(void* const* d_in, const int* in_sizes, int n_in,
                              void* d_out, int out_size, void* d_ws, size_t ws_size,
                              hipStream_t stream) {
  const float* x   = (const float*)d_in[0];
  const int*   nbr = (const int*)  d_in[1];
  const float* W1  = (const float*)d_in[2];
  const float* g1  = (const float*)d_in[3];
  const float* b1  = (const float*)d_in[4];
  const float* rm1 = (const float*)d_in[5];
  const float* rv1 = (const float*)d_in[6];
  const float* W2  = (const float*)d_in[7];
  const float* g2  = (const float*)d_in[8];
  const float* b2  = (const float*)d_in[9];
  const float* rm2 = (const float*)d_in[10];
  const float* rv2 = (const float*)d_in[11];

  unsigned short* xbf = (unsigned short*)d_ws;         // N*CH bf16 = 25.6 MB
  unsigned short* mid = xbf + (size_t)N_SITES * CH;    // N*CH bf16 = 25.6 MB
  float* out = (float*)d_out;

  const int cvt_blocks  = ((N_SITES * CH) / 8 + 255) / 256;       // 6250
  const int conv_blocks = (N_WTILES + WAVES - 1) / WAVES;         // 782

  cvt_bf16<<<cvt_blocks, 256, 0, stream>>>(x, xbf);
  subm_conv<false><<<conv_blocks, BLOCK_THREADS, 0, stream>>>(
      xbf, nullptr, nbr, W1, g1, b1, rm1, rv1, mid, nullptr);
  subm_conv<true><<<conv_blocks, BLOCK_THREADS, 0, stream>>>(
      mid, x, nbr, W2, g2, b2, rm2, rv2, nullptr, out);
}
